// MultiHeadAttentionSelf_74156905333157
// MI455X (gfx1250) — matrix operand order repaired and checked by reading, not compiled, not run on hardware
//
#include <hip/hip_runtime.h>

// ---------------------------------------------------------------- constants
constexpr int B_   = 4;
constexpr int QL_  = 1024;
constexpr int KVL_ = 4096;
constexpr int D_   = 2048;
constexpr int H_   = 16;
constexpr int DH_  = 128;
constexpr float SOFTSCALE = 0.08838834764831845f;   // DH^-0.5  (ref applies DH^-0.25 to q and k)

// ---------------------------------------------------------------- types
typedef __attribute__((ext_vector_type(16))) __bf16 v16bf;
typedef __attribute__((ext_vector_type(8)))  __bf16 bf16x8;
typedef __attribute__((ext_vector_type(8)))  float  v8f;
typedef __attribute__((ext_vector_type(4)))  unsigned int u32x4;
typedef __attribute__((ext_vector_type(4)))  int i32x4;
typedef __attribute__((ext_vector_type(8)))  int i32x8;

union Frag16 { v16bf v; bf16x8 h[2]; };

#if __has_builtin(__builtin_amdgcn_tensor_load_to_lds)
#define USE_TDM 1
#else
#define USE_TDM 0
#endif

__device__ __forceinline__ __bf16 f2bf(float f){
    unsigned u = __builtin_bit_cast(unsigned, f);
    unsigned r = u + 0x7fffu + ((u >> 16) & 1u);    // round-to-nearest-even
    unsigned short hs = (unsigned short)(r >> 16);
    return __builtin_bit_cast(__bf16, hs);
}

#if USE_TDM
// TDM load of a 2D bf16 tile [rows][width] (global row stride strideElems) into LDS,
// with TDM padding producing LDS row stride = width + 8*(padAmountCode+... ) elements.
// padIntervalCode: row length in DWORDs as code (3 => 16 dw = 32 bf16, 5 => 64 dw = 128 bf16)
// padAmountCode 3 => 4 DWORDs = 8 bf16 pad per row.
__device__ __forceinline__ void tdm_load_tile_bf16(const __bf16* gptr, void* ldsPtr,
                                                   int width, int rows, int strideElems,
                                                   int padIntervalCode, int padAmountCode){
    unsigned lds_addr = (unsigned)(unsigned long long)ldsPtr;   // LDS byte offset (low 32 bits)
    unsigned long long ga = (unsigned long long)gptr;
    u32x4 g0;
    g0[0] = 1u;                                            // count=1, user descriptor
    g0[1] = lds_addr;                                      // [63:32] lds_addr
    g0[2] = (unsigned)(ga & 0xffffffffu);                  // [95:64] global addr lo
    g0[3] = (unsigned)((ga >> 32) & 0x01ffffffu) | (2u << 30); // addr hi, type=2 ("image")
    i32x8 g1;
    g1[0] = (1 << 16)                      // data_size = 2 bytes
          | (1 << 20)                      // pad_enable
          | (padIntervalCode << 22)        // pad_interval
          | (padAmountCode  << 25);        // pad_amount
    g1[1] = width << 16;                   // tensor_dim0 (lo16 at bits 63:48)
    g1[2] = rows  << 16;                   // tensor_dim1 (lo16 at bits 95:80)
    g1[3] = width << 16;                   // tile_dim0   (bits 127:112)
    g1[4] = rows;                          // tile_dim1   (bits 143:128)
    g1[5] = strideElems;                   // tensor_dim0_stride lo32
    g1[6] = 0;
    g1[7] = 0;
    i32x4 gz = {0, 0, 0, 0};
#if defined(__clang_major__) && (__clang_major__ >= 23)
    i32x8 gz8 = {0, 0, 0, 0, 0, 0, 0, 0};
    __builtin_amdgcn_tensor_load_to_lds(g0, g1, gz, gz, gz8, 0);
#else
    __builtin_amdgcn_tensor_load_to_lds(g0, g1, gz, gz, 0);
#endif
}
#endif

// A-fragment (16x32 bf16, M x K) from row-major LDS tile.
__device__ __forceinline__ v16bf lds_a_frag(const __bf16* lds, int rowBase, int colBase, int stride){
    const int lane = threadIdx.x & 31;
    const int row  = rowBase + (lane & 15);
    const int kh   = lane >> 4;
    const __bf16* p = lds + row*stride + colBase + kh*8;
    Frag16 f;
    f.h[0] = *(const bf16x8*)(p);
    f.h[1] = *(const bf16x8*)(p + 16);
    return f.v;
}

// B-fragment (32x16 bf16, K x N) from an LDS tile stored as [N][K].
__device__ __forceinline__ v16bf lds_b_frag(const __bf16* lds, int nBase, int kBase, int stride){
    const int lane = threadIdx.x & 31;
    const int n    = nBase + (lane & 15);
    const int kh   = lane >> 4;
    const __bf16* p = lds + n*stride + kBase + kh*8;
    Frag16 f;
    f.h[0] = *(const bf16x8*)(p);
    f.h[1] = *(const bf16x8*)(p + 16);
    return f.v;
}

__device__ __forceinline__ v8f wmma_bf16(v16bf a, v16bf b, v8f c){
    return __builtin_amdgcn_wmma_f32_16x16x32_bf16(false, a, false, b, (short)0, c, false, false);
}

// ---------------------------------------------------------------- f32 -> bf16
__global__ void f32_to_bf16_kernel(const float* __restrict__ src, __bf16* __restrict__ dst,
                                   long long n){
    long long i = (long long)blockIdx.x * blockDim.x + threadIdx.x;
    long long stride = (long long)gridDim.x * blockDim.x;
    for (; i < n; i += stride) dst[i] = f2bf(src[i]);
}

// ---------------------------------------------------------------- tiled WMMA GEMM
// C[M,N] = A[M,K] @ Bm[K,N] (+bias).  128x128 block tile, 8 waves, K-step 32.
// A tile fetched by the Tensor Data Mover (double-buffered, TENSORcnt);
// B tile fetched by vector loads + LDS transpose store.
__global__ __launch_bounds__(256)
void gemm_bf16_128x128(const __bf16* __restrict__ A, const __bf16* __restrict__ Bm,
                       const float* __restrict__ bias,
                       float* __restrict__ Cf, __bf16* __restrict__ Cb,
                       int M, int N, int K, int cacheMode){
    __shared__ __bf16 sA [2][128*40];  // row-major [128 m][32 k], stride 40 (TDM pad)
    __shared__ __bf16 sBT[128*40];     // transposed [128 n][32 k], stride 40

    const int tid  = threadIdx.x;
    const int wave = tid >> 5;
    const int wm   = (wave >> 1) * 32;
    const int wn   = (wave & 1) * 64;
    const int m0   = blockIdx.y * 128;
    const int n0   = blockIdx.x * 128;

    v8f acc[2][4];
    #pragma unroll
    for (int i = 0; i < 2; i++)
        #pragma unroll
        for (int j = 0; j < 4; j++){
            v8f z = {0.f,0.f,0.f,0.f,0.f,0.f,0.f,0.f};
            acc[i][j] = z;
        }

    const int nk = K / 32;
#if USE_TDM
    if (wave == 0)   // prologue: A tile 0, [128 rows][32 cols], row stride K
        tdm_load_tile_bf16(A + (size_t)m0*K, (void*)sA[0], 32, 128, K, 3, 3);
#endif
    for (int kt = 0; kt < nk; ++kt){
        const int k0 = kt * 32;
#if !USE_TDM
        #pragma unroll
        for (int i = 0; i < 2; i++){
            int chunk = tid + i*256;
            int row = chunk >> 2, coff = (chunk & 3) * 8;
            bf16x8 v = *(const bf16x8*)(A + (size_t)(m0 + row)*K + k0 + coff);
            *(bf16x8*)(sA[kt & 1] + row*40 + coff) = v;
        }
#endif
        // B tile: 32x128, stored transposed
        #pragma unroll
        for (int i = 0; i < 2; i++){
            int chunk = tid + i*256;
            int kr = chunk >> 4, noff = (chunk & 15) * 8;
            bf16x8 v = *(const bf16x8*)(Bm + (size_t)(k0 + kr)*N + n0 + noff);
            #pragma unroll
            for (int j = 0; j < 8; j++) sBT[(noff + j)*40 + kr] = v[j];
        }
        if (kt + 1 < nk)
            __builtin_prefetch(Bm + (size_t)(k0 + 32 + (tid >> 3))*N + n0 + (tid & 7)*16, 0, 0);
#if USE_TDM
        if (wave == 0){
            if (kt + 1 < nk){   // kick off next A tile, then wait for current one
                tdm_load_tile_bf16(A + (size_t)m0*K + k0 + 32, (void*)sA[(kt + 1) & 1],
                                   32, 128, K, 3, 3);
                __builtin_amdgcn_s_wait_tensorcnt((short)1);
            } else {
                __builtin_amdgcn_s_wait_tensorcnt((short)0);
            }
        }
#endif
        __syncthreads();

        const __bf16* sAcur = sA[kt & 1];
        v16bf afrag[2], bfrag[4];
        #pragma unroll
        for (int mi = 0; mi < 2; mi++) afrag[mi] = lds_a_frag(sAcur, wm + mi*16, 0, 40);
        #pragma unroll
        for (int ni = 0; ni < 4; ni++) bfrag[ni] = lds_b_frag(sBT, wn + ni*16, 0, 40);
        #pragma unroll
        for (int mi = 0; mi < 2; mi++)
            #pragma unroll
            for (int ni = 0; ni < 4; ni++)
                acc[mi][ni] = wmma_bf16(afrag[mi], bfrag[ni], acc[mi][ni]);
        __syncthreads();
    }

    const int lane   = tid & 31;
    const int halfId = lane >> 4;
    const int nloc   = lane & 15;
    #pragma unroll
    for (int mi = 0; mi < 2; mi++)
        #pragma unroll
        for (int ni = 0; ni < 4; ni++)
            #pragma unroll
            for (int r = 0; r < 8; r++){
                int gr = m0 + wm + mi*16 + r + 8*halfId;
                int gc = n0 + wn + ni*16 + nloc;
                float v = acc[mi][ni][r];
                if (bias) v += bias[gc];
                if (Cf){
                    int orow = gr;
                    if (cacheMode)  // scatter k_new/v_new into cache tail rows
                        orow = (gr / QL_)*KVL_ + (KVL_ - QL_) + (gr % QL_);
                    Cf[(size_t)orow*N + gc] = v;
                }
                if (Cb) Cb[(size_t)gr*N + gc] = f2bf(v);
            }
}

// ---------------------------------------------------------------- flash attention
// grid = (QL/64, B*H), block = 128 (4 waves, 16 q-rows each). 32-key KV steps.
// K tiles streamed by TDM (double-buffered); V tiles vector-loaded + LDS-transposed.
__global__ __launch_bounds__(128)
void flash_attn_bf16(const __bf16* __restrict__ Q, const __bf16* __restrict__ Kc,
                     const __bf16* __restrict__ Vc, __bf16* __restrict__ WV){
    __shared__ __bf16 sQ [64*136];     // q tile [64 q][128 dh]
    __shared__ __bf16 sK [2][32*136];  // k tile [32 kv][128 dh] (TDM pad -> stride 136)
    __shared__ __bf16 sVT[128*40];     // v tile transposed [128 dh][32 kv]
    __shared__ __bf16 sP [64*40];      // P tiles, per-wave 16-row regions

    const int tid    = threadIdx.x;
    const int bh     = blockIdx.y;
    const int b      = bh >> 4;        // H=16
    const int h      = bh & 15;
    const int qbase  = blockIdx.x * 64;
    const int wave   = tid >> 5;
    const int qw     = wave * 16;
    const int lane   = tid & 31;
    const int halfId = lane >> 4;
    const int nloc   = lane & 15;

    const __bf16* kBase = Kc + (size_t)b*KVL_*D_ + h*DH_;
    const __bf16* vBase = Vc + (size_t)b*KVL_*D_ + h*DH_;

    // load Q tile (64x128)
    #pragma unroll
    for (int i = 0; i < 8; i++){
        int chunk = tid + i*128;
        int row = chunk >> 4, coff = (chunk & 15) * 8;
        bf16x8 v = *(const bf16x8*)(Q + (size_t)(b*QL_ + qbase + row)*D_ + h*DH_ + coff);
        *(bf16x8*)(sQ + row*136 + coff) = v;
    }

    v8f o[8];
    #pragma unroll
    for (int dt = 0; dt < 8; dt++){
        v8f z = {0.f,0.f,0.f,0.f,0.f,0.f,0.f,0.f};
        o[dt] = z;
    }
    float mrow[8], lrow[8];
    #pragma unroll
    for (int r = 0; r < 8; r++){ mrow[r] = -3.0e38f; lrow[r] = 0.f; }

    const int nt = KVL_ / 32;
#if USE_TDM
    if (wave == 0)   // prologue: K tile 0, [32 rows][128 cols], row stride D
        tdm_load_tile_bf16(kBase, (void*)sK[0], DH_, 32, D_, 5, 3);
#endif
    for (int t = 0; t < nt; ++t){
        const int kv0 = t * 32;
        // V tile: vector load + transpose into sVT
        #pragma unroll
        for (int i = 0; i < 4; i++){
            int chunk = tid + i*128;
            int row = chunk >> 4, coff = (chunk & 15) * 8;
            bf16x8 vv = *(const bf16x8*)(vBase + (size_t)(kv0 + row)*D_ + coff);
            #pragma unroll
            for (int j = 0; j < 8; j++) sVT[(coff + j)*40 + row] = vv[j];
        }
#if USE_TDM
        if (wave == 0){
            if (t + 1 < nt){
                tdm_load_tile_bf16(kBase + (size_t)(kv0 + 32)*D_, (void*)sK[(t + 1) & 1],
                                   DH_, 32, D_, 5, 3);
                __builtin_amdgcn_s_wait_tensorcnt((short)1);
            } else {
                __builtin_amdgcn_s_wait_tensorcnt((short)0);
            }
        }
#else
        #pragma unroll
        for (int i = 0; i < 4; i++){
            int chunk = tid + i*128;
            int row = chunk >> 4, coff = (chunk & 15) * 8;
            bf16x8 kv = *(const bf16x8*)(kBase + (size_t)(kv0 + row)*D_ + coff);
            *(bf16x8*)(sK[t & 1] + row*136 + coff) = kv;
        }
#endif
        __syncthreads();

        const __bf16* sKcur = sK[t & 1];
        // S[16 q][32 kv] = Q @ K^T  (4 dh-chunks of 32)
        v8f s[2];
        {
            v8f z = {0.f,0.f,0.f,0.f,0.f,0.f,0.f,0.f};
            s[0] = z; s[1] = z;
        }
        #pragma unroll
        for (int kc = 0; kc < 4; kc++){
            v16bf aq = lds_a_frag(sQ, qw, kc*32, 136);
            #pragma unroll
            for (int ntl = 0; ntl < 2; ntl++){
                v16bf bk = lds_b_frag(sKcur, ntl*16, kc*32, 136);
                s[ntl] = wmma_bf16(aq, bk, s[ntl]);
            }
        }

        // scale + analytic causal mask + online softmax (row reduce in 16-lane half)
        float alpha[8];
        #pragma unroll
        for (int r = 0; r < 8; r++){
            int qr = qbase + qw + 8*halfId + r;
            float s0 = s[0][r]*SOFTSCALE + (((kv0 + nloc)      > (KVL_-QL_) + qr) ? -1e9f : 0.f);
            float s1 = s[1][r]*SOFTSCALE + (((kv0 + 16 + nloc) > (KVL_-QL_) + qr) ? -1e9f : 0.f);
            float mx = fmaxf(s0, s1);
            #pragma unroll
            for (int off = 8; off >= 1; off >>= 1) mx = fmaxf(mx, __shfl_xor(mx, off, 32));
            float mnew = fmaxf(mrow[r], mx);
            float a  = __expf(mrow[r] - mnew);
            float p0 = __expf(s0 - mnew);
            float p1 = __expf(s1 - mnew);
            float rs = p0 + p1;
            #pragma unroll
            for (int off = 8; off >= 1; off >>= 1) rs += __shfl_xor(rs, off, 32);
            lrow[r] = lrow[r]*a + rs;
            mrow[r] = mnew;
            alpha[r] = a;
            s[0][r] = p0; s[1][r] = p1;
        }
        #pragma unroll
        for (int dt = 0; dt < 8; dt++)
            #pragma unroll
            for (int r = 0; r < 8; r++) o[dt][r] *= alpha[r];

        // re-layout P (C-layout) -> A-fragment layout via per-wave LDS region
        __bf16* myP = sP + qw*40;
        #pragma unroll
        for (int r = 0; r < 8; r++){
            myP[(r + 8*halfId)*40 + nloc]      = f2bf(s[0][r]);
            myP[(r + 8*halfId)*40 + 16 + nloc] = f2bf(s[1][r]);
        }
        v16bf ap = lds_a_frag(myP, 0, 0, 40);
        #pragma unroll
        for (int dt = 0; dt < 8; dt++){
            v16bf bv = lds_b_frag(sVT, dt*16, 0, 40);
            o[dt] = wmma_bf16(ap, bv, o[dt]);
        }
        __syncthreads();
    }

    float rinv[8];
    #pragma unroll
    for (int r = 0; r < 8; r++) rinv[r] = 1.0f / lrow[r];
    #pragma unroll
    for (int dt = 0; dt < 8; dt++)
        #pragma unroll
        for (int r = 0; r < 8; r++){
            int row = b*QL_ + qbase + qw + 8*halfId + r;
            int col = h*DH_ + dt*16 + nloc;
            WV[(size_t)row*D_ + col] = f2bf(o[dt][r] * rinv[r]);
        }
}

// ---------------------------------------------------------------- launch
extern "C" void kernel_launch(void* const* d_in, const int* in_sizes, int n_in,
                              void* d_out, int out_size, void* d_ws, size_t ws_size,
                              hipStream_t stream){
    (void)in_sizes; (void)n_in; (void)out_size; (void)ws_size;

    const float* x   = (const float*)d_in[0];
    const float* kca = (const float*)d_in[1];
    const float* vca = (const float*)d_in[2];
    const float* Wq  = (const float*)d_in[3];
    const float* bq  = (const float*)d_in[4];
    const float* Wk  = (const float*)d_in[5];
    const float* Wv  = (const float*)d_in[6];
    const float* bv  = (const float*)d_in[7];
    const float* Wo  = (const float*)d_in[8];
    const float* bo  = (const float*)d_in[9];
    // d_in[10] = mask: causal tail mask computed analytically in-kernel

    const size_t nBQLD  = (size_t)B_*QL_*D_;     //  8,388,608
    const size_t nBKVLD = (size_t)B_*KVL_*D_;    // 33,554,432
    const size_t nDD    = (size_t)D_*D_;         //  4,194,304

    float* out = (float*)d_out;
    float* kc  = out + nBQLD;
    float* vc  = kc  + nBKVLD;

    __bf16* p = (__bf16*)d_ws;
    __bf16* x_bf  = p; p += nBQLD;
    __bf16* wq_bf = p; p += nDD;
    __bf16* wk_bf = p; p += nDD;
    __bf16* wv_bf = p; p += nDD;
    __bf16* wo_bf = p; p += nDD;
    __bf16* q_bf  = p; p += nBQLD;
    __bf16* kc_bf = p; p += nBKVLD;
    __bf16* vc_bf = p; p += nBKVLD;
    __bf16* av_bf = p; p += nBQLD;   // attention output (pre-Wo)

    // cache heads (tails overwritten by K/V projection GEMMs below)
    hipMemcpyAsync(kc, kca, nBKVLD*sizeof(float), hipMemcpyDeviceToDevice, stream);
    hipMemcpyAsync(vc, vca, nBKVLD*sizeof(float), hipMemcpyDeviceToDevice, stream);

    f32_to_bf16_kernel<<<2048, 256, 0, stream>>>(x,  x_bf,  (long long)nBQLD);
    f32_to_bf16_kernel<<<1024, 256, 0, stream>>>(Wq, wq_bf, (long long)nDD);
    f32_to_bf16_kernel<<<1024, 256, 0, stream>>>(Wk, wk_bf, (long long)nDD);
    f32_to_bf16_kernel<<<1024, 256, 0, stream>>>(Wv, wv_bf, (long long)nDD);
    f32_to_bf16_kernel<<<1024, 256, 0, stream>>>(Wo, wo_bf, (long long)nDD);

    dim3 ggrid(D_/128, (B_*QL_)/128);    // (16, 32)
    // q = x@Wq + bq  -> bf16 only (consumed by attention)
    gemm_bf16_128x128<<<ggrid, 256, 0, stream>>>(x_bf, wq_bf, bq, nullptr, q_bf,
                                                 B_*QL_, D_, D_, 0);
    // k_new = x@Wk (no bias) -> f32 scatter into kc tail
    gemm_bf16_128x128<<<ggrid, 256, 0, stream>>>(x_bf, wk_bf, nullptr, kc, nullptr,
                                                 B_*QL_, D_, D_, 1);
    // v_new = x@Wv + bv -> f32 scatter into vc tail
    gemm_bf16_128x128<<<ggrid, 256, 0, stream>>>(x_bf, wv_bf, bv, vc, nullptr,
                                                 B_*QL_, D_, D_, 1);

    // bf16 copies of completed caches for the attention GEMMs
    f32_to_bf16_kernel<<<4096, 256, 0, stream>>>(kc, kc_bf, (long long)nBKVLD);
    f32_to_bf16_kernel<<<4096, 256, 0, stream>>>(vc, vc_bf, (long long)nBKVLD);

    flash_attn_bf16<<<dim3(QL_/64, B_*H_), 128, 0, stream>>>(q_bf, kc_bf, vc_bf, av_bf);

    // out = wv@Wo + bo (f32)
    gemm_bf16_128x128<<<ggrid, 256, 0, stream>>>(av_bf, wo_bf, bo, out, nullptr,
                                                 B_*QL_, D_, D_, 0);
}
